// TrajPreAttnAvgLongUser_31241592111495
// MI455X (gfx1250) — matrix operand order repaired and checked by reading, not compiled, not run on hardware
//
#include <hip/hip_runtime.h>
#include <hip/hip_bf16.h>

// ---------------- problem constants (match reference) ----------------
#define S_LEN  1024
#define T_LEN  512
#define G_CNT  1024
#define HID    512
#define H3     1536     // 3*HID
#define DLOC   512
#define DCLU   64
#define DTIM   32
#define DUID   64
#define INP    608      // DLOC+DCLU+DTIM
#define CAT    1088     // 2*HID + DUID
#define LOC_N  50000    // 3125 * 16

typedef __attribute__((ext_vector_type(16))) __bf16 v16bf;
typedef __attribute__((ext_vector_type(8)))  float  v8f;

__device__ __forceinline__ float sigmoid_f(float x) { return 1.0f / (1.0f + __expf(-x)); }

// ---------------------------------------------------------------------
// WMMA fragment loaders for NT GEMM:  C[m][n] = sum_k A[m][k] * B[n][k]
// A is 16x32 (MxK): lane l (l&15 = M row), halves 0..15 = K kb..kb+15
// with kb = (l>>4)*16.
// B operand is the 32x16 KxN matrix = Bmem[n][k]: lane l holds column
// n = l&15 over 16 contiguous K starting at (l>>4)*16.
// Both patterns are contiguous float4 loads converted to bf16 in regs.
// ---------------------------------------------------------------------
__device__ __forceinline__ v16bf load_a_frag(const float* __restrict__ A, int ldk,
                                             int r0, int k0, int lane) {
  const float* p = A + (size_t)(r0 + (lane & 15)) * ldk + k0 + ((lane >> 4) << 4);
  float4 x0 = *(const float4*)(p + 0);
  float4 x1 = *(const float4*)(p + 4);
  float4 y0 = *(const float4*)(p + 8);
  float4 y1 = *(const float4*)(p + 12);
  v16bf f;
  f[0]  = (__bf16)x0.x; f[1]  = (__bf16)x0.y; f[2]  = (__bf16)x0.z; f[3]  = (__bf16)x0.w;
  f[4]  = (__bf16)x1.x; f[5]  = (__bf16)x1.y; f[6]  = (__bf16)x1.z; f[7]  = (__bf16)x1.w;
  f[8]  = (__bf16)y0.x; f[9]  = (__bf16)y0.y; f[10] = (__bf16)y0.z; f[11] = (__bf16)y0.w;
  f[12] = (__bf16)y1.x; f[13] = (__bf16)y1.y; f[14] = (__bf16)y1.z; f[15] = (__bf16)y1.w;
  return f;
}

__device__ __forceinline__ v16bf load_b_frag(const float* __restrict__ B, int ldk,
                                             int n0, int k0, int lane) {
  const float* p = B + (size_t)(n0 + (lane & 15)) * ldk + k0 + ((lane >> 4) << 4);
  float4 x0 = *(const float4*)(p + 0);
  float4 x1 = *(const float4*)(p + 4);
  float4 x2 = *(const float4*)(p + 8);
  float4 x3 = *(const float4*)(p + 12);
  v16bf f;
  f[0]  = (__bf16)x0.x; f[1]  = (__bf16)x0.y; f[2]  = (__bf16)x0.z; f[3]  = (__bf16)x0.w;
  f[4]  = (__bf16)x1.x; f[5]  = (__bf16)x1.y; f[6]  = (__bf16)x1.z; f[7]  = (__bf16)x1.w;
  f[8]  = (__bf16)x2.x; f[9]  = (__bf16)x2.y; f[10] = (__bf16)x2.z; f[11] = (__bf16)x2.w;
  f[12] = (__bf16)x3.x; f[13] = (__bf16)x3.y; f[14] = (__bf16)x3.z; f[15] = (__bf16)x3.w;
  return f;
}

// ---------------------------------------------------------------------
// Generic NT GEMM, bf16 WMMA, f32 accumulate.
//   grid.x = N/16   (one 16-col strip of B per block -> B streamed once
//                    from HBM when grid.y == 1; A stays L2-resident)
//   grid.y = M/512; block = 256 threads = 8 waves; each wave owns
//   GEMM_MT=4 m-tiles (64 rows) so a block covers exactly 512 rows.
// LAUNCH CONTRACT (all call sites obey): M % 512 == 0 covered exactly by
// grid.y, N % 16 == 0, K % 32 == 0. No runtime guards -> branchless inner
// loop, accumulators stay pinned in WMMA registers (no exec-mask phi
// copies as in the previous revision).
// act: 0 = none, 1 = tanh. bias may be nullptr.
// ---------------------------------------------------------------------
#define GEMM_MT 4

__global__ void __launch_bounds__(256)
gemm_nt_bf16(const float* __restrict__ A, const float* __restrict__ B,
             const float* __restrict__ bias, float* __restrict__ C,
             int M, int N, int K, int act) {
  const int lane  = threadIdx.x & 31;
  const int wave  = threadIdx.x >> 5;
  const int n0    = blockIdx.x << 4;
  const int mBase = blockIdx.y * (8 * GEMM_MT * 16) + wave * (GEMM_MT * 16);
  (void)M;  // coverage guaranteed by launch contract

  v8f acc[GEMM_MT] = {};

  const int brow = n0 + (lane & 15);
  for (int k0 = 0; k0 < K; k0 += 32) {
    // prefetch next K-block of this weight row (global_prefetch_b8)
    if (k0 + 32 < K)
      __builtin_prefetch(B + (size_t)brow * K + k0 + 32 + ((lane >> 4) << 4), 0, 1);
    const v16bf bf = load_b_frag(B, K, n0, k0, lane);
#pragma unroll
    for (int t = 0; t < GEMM_MT; ++t) {
      const v16bf af = load_a_frag(A, K, mBase + t * 16, k0, lane);
      acc[t] = __builtin_amdgcn_wmma_f32_16x16x32_bf16(
          false, af, false, bf, (short)0, acc[t], false, false);
    }
  }

  // C/D layout: lane l -> col = l&15, rows rOff..rOff+7 with rOff = (l>>4)*8
  const int col  = n0 + (lane & 15);
  const int rOff = (lane >> 4) << 3;
  const float bv = bias ? bias[col] : 0.0f;
#pragma unroll
  for (int t = 0; t < GEMM_MT; ++t) {
    const int r0 = mBase + t * 16;
#pragma unroll
    for (int v = 0; v < 8; ++v) {
      float val = acc[t][v] + bv;
      if (act == 1) val = tanhf(val);
      C[(size_t)(r0 + rOff + v) * N + col] = val;
    }
  }
}

// ---------------------------------------------------------------------
// Small helper kernels
// ---------------------------------------------------------------------
__global__ void __launch_bounds__(1024)
scan_offsets_kernel(const int* __restrict__ cnt, int* __restrict__ offs) {
  __shared__ int s[G_CNT];
  const int i = threadIdx.x;
  s[i] = cnt[i];
  __syncthreads();
  for (int d = 1; d < G_CNT; d <<= 1) {
    int v = (i >= d) ? s[i - d] : 0;
    __syncthreads();
    s[i] += v;
    __syncthreads();
  }
  offs[i] = s[i] - cnt[i];   // exclusive prefix sum
}

__global__ void gather_x_kernel(const int* __restrict__ loc, const int* __restrict__ clu,
                                const int* __restrict__ tim,
                                const float* __restrict__ eloc, const float* __restrict__ eclu,
                                const float* __restrict__ etim, float* __restrict__ x) {
  const int s = blockIdx.x;
  const int l = loc[s], c = clu[s], t = tim[s];
  float* row = x + (size_t)s * INP;
  for (int d = threadIdx.x; d < DLOC; d += blockDim.x) row[d]               = eloc[(size_t)l * DLOC + d];
  for (int d = threadIdx.x; d < DCLU; d += blockDim.x) row[DLOC + d]        = eclu[(size_t)c * DCLU + d];
  for (int d = threadIdx.x; d < DTIM; d += blockDim.x) row[DLOC + DCLU + d] = etim[(size_t)t * DTIM + d];
}

__global__ void hist_feat_kernel(const int* __restrict__ hloc, const int* __restrict__ hclu,
                                 const int* __restrict__ htim, const int* __restrict__ cnt,
                                 const int* __restrict__ offs,
                                 const float* __restrict__ eloc, const float* __restrict__ eclu,
                                 const float* __restrict__ etim, float* __restrict__ pre) {
  const int g = blockIdx.x;
  const int c = cnt[g], o = offs[g];
  const float inv = 1.0f / (float)c;
  float* row = pre + (size_t)g * INP;
  for (int d = threadIdx.x; d < DLOC; d += blockDim.x) {
    float a = 0.0f;
    for (int j = 0; j < c; ++j) a += eloc[(size_t)hloc[o + j] * DLOC + d];
    row[d] = a * inv;
  }
  for (int d = threadIdx.x; d < DCLU; d += blockDim.x) {
    float a = 0.0f;
    for (int j = 0; j < c; ++j) a += eclu[(size_t)hclu[o + j] * DCLU + d];
    row[DLOC + d] = a * inv;
  }
  const int tf = htim[o];
  for (int d = threadIdx.x; d < DTIM; d += blockDim.x)
    row[DLOC + DCLU + d] = etim[(size_t)tf * DTIM + d];
}

__global__ void transpose_kernel(const float* __restrict__ in, float* __restrict__ out,
                                 int R, int C) {   // out[C][R] = in[R][C]^T
  __shared__ float tile[32][33];
  const int rb = blockIdx.y * 32, cb = blockIdx.x * 32;
  const int tx = threadIdx.x & 31, ty = threadIdx.x >> 5;
  for (int i = ty; i < 32; i += 8) {
    int r = rb + i, c = cb + tx;
    tile[i][tx] = (r < R && c < C) ? in[(size_t)r * C + c] : 0.0f;
  }
  __syncthreads();
  for (int i = ty; i < 32; i += 8) {
    int r = cb + i, c = rb + tx;
    if (r < C && c < R) out[(size_t)r * R + c] = tile[tx][i];
  }
}

__global__ void softmax_rows_kernel(float* __restrict__ a, int cols) {
  __shared__ float red[256];
  const int r = blockIdx.x, tid = threadIdx.x;
  float* row = a + (size_t)r * cols;
  float m = -3.4e38f;
  for (int c = tid; c < cols; c += 256) m = fmaxf(m, row[c]);
  red[tid] = m; __syncthreads();
  for (int s = 128; s > 0; s >>= 1) { if (tid < s) red[tid] = fmaxf(red[tid], red[tid + s]); __syncthreads(); }
  m = red[0]; __syncthreads();
  float sum = 0.0f;
  for (int c = tid; c < cols; c += 256) { float e = __expf(row[c] - m); row[c] = e; sum += e; }
  red[tid] = sum; __syncthreads();
  for (int s = 128; s > 0; s >>= 1) { if (tid < s) red[tid] += red[tid + s]; __syncthreads(); }
  const float inv = 1.0f / red[0];
  for (int c = tid; c < cols; c += 256) row[c] *= inv;
}

__global__ void log_softmax_rows_kernel(float* __restrict__ y, int cols) {
  __shared__ float red[256];
  const int r = blockIdx.x, tid = threadIdx.x;
  float* row = y + (size_t)r * cols;
  float m = -3.4e38f;
  for (int c = tid; c < cols; c += 256) m = fmaxf(m, row[c]);
  red[tid] = m; __syncthreads();
  for (int s = 128; s > 0; s >>= 1) { if (tid < s) red[tid] = fmaxf(red[tid], red[tid + s]); __syncthreads(); }
  m = red[0]; __syncthreads();
  float sum = 0.0f;
  for (int c = tid; c < cols; c += 256) sum += __expf(row[c] - m);
  red[tid] = sum; __syncthreads();
  for (int s = 128; s > 0; s >>= 1) { if (tid < s) red[tid] += red[tid + s]; __syncthreads(); }
  const float lse = m + logf(red[0]);
  for (int c = tid; c < cols; c += 256) row[c] -= lse;
}

__global__ void concat_out_kernel(const float* __restrict__ q, const float* __restrict__ ctx,
                                  const float* __restrict__ euid, const int* __restrict__ uid,
                                  float* __restrict__ out) {
  const int r = blockIdx.x;
  const int u = uid[0];
  float* row = out + (size_t)r * CAT;
  for (int d = threadIdx.x; d < HID; d += blockDim.x) {
    row[d]       = q[(size_t)r * HID + d];
    row[HID + d] = ctx[(size_t)r * HID + d];
  }
  for (int d = threadIdx.x; d < DUID; d += blockDim.x)
    row[2 * HID + d] = euid[(size_t)u * DUID + d];
}

__global__ void ws_init_kernel(int* bar) { if (threadIdx.x < 2) bar[threadIdx.x] = 0; }

// ---------------------------------------------------------------------
// Persistent multi-WGP GRU.
// 16 blocks; block b pins a bf16 copy of w_hh rows [b*96, b*96+96) in LDS
// (96KB << 320KB CDNA5 LDS) and computes its 96 entries of gh = h @ w_hhᵀ
// per step using WMMA (A-frag: row 0 = h, rows 1..15 = 0). One global
// sense-reversing barrier per step; gh double-buffered; every block then
// redundantly (deterministically) computes the identical gate update.
// ---------------------------------------------------------------------
#define GRU_BLOCKS  16
#define GRU_THREADS 192                          // 6 waves * 16 cols = 96
#define NSLICE      96
#define GRU_SMEM    ((size_t)NSLICE * HID * 2 + (size_t)HID * 4 + (size_t)HID * 2)

__device__ void grid_barrier(int* bar, int nblocks) {
  __threadfence();
  __syncthreads();
  if (threadIdx.x == 0) {
    volatile int* vgen = bar + 1;
    int g = *vgen;
    if (atomicAdd(bar, 1) == nblocks - 1) {
      bar[0] = 0;
      __threadfence();
      atomicAdd(bar + 1, 1);
    } else {
      while (*vgen == g) __builtin_amdgcn_s_sleep(2);
    }
  }
  __syncthreads();
}

__global__ void __launch_bounds__(GRU_THREADS)
gru_persistent(const float* __restrict__ gx, const float* __restrict__ w_hh,
               const float* __restrict__ b_hh, float* __restrict__ outs,
               float* __restrict__ ghbuf, int* bar) {
  extern __shared__ unsigned char smem[];
  __bf16* sw  = (__bf16*)smem;                                       // 96x512 bf16 weights
  float*  sh  = (float*)(smem + (size_t)NSLICE * HID * 2);           // h (f32)
  __bf16* shb = (__bf16*)(smem + (size_t)NSLICE * HID * 2 + HID * 4);// h (bf16)

  const int tid   = threadIdx.x;
  const int lane  = tid & 31;
  const int wave  = tid >> 5;            // 0..5
  const int nBase = blockIdx.x * NSLICE;
  const int nloc  = wave << 4;

  for (int i = tid; i < NSLICE * HID; i += GRU_THREADS) {
    int n = i >> 9, k = i & (HID - 1);
    sw[i] = (__bf16)w_hh[(size_t)(nBase + n) * HID + k];
  }
  for (int i = tid; i < HID; i += GRU_THREADS) { sh[i] = 0.0f; shb[i] = (__bf16)0.0f; }
  __syncthreads();

  const float bb = b_hh[nBase + nloc + (lane & 15)];

  for (int t = 0; t < S_LEN; ++t) {
    float* gh = ghbuf + (t & 1) * H3;
    v8f acc = {};
#pragma unroll 4
    for (int kk = 0; kk < HID; kk += 32) {
      v16bf af = {};
      if ((lane & 15) == 0) {              // only row 0 of A carries h
        const int kb = kk + ((lane >> 4) << 4);
#pragma unroll
        for (int i = 0; i < 8; ++i) { af[i] = shb[kb + i]; af[8 + i] = shb[kb + 8 + i]; }
      }
      const v16bf* bp = (const v16bf*)(sw + (size_t)(nloc + (lane & 15)) * HID +
                                       kk + ((lane >> 4) << 4));
      acc = __builtin_amdgcn_wmma_f32_16x16x32_bf16(
          false, af, false, *bp, (short)0, acc, false, false);
    }
    if (lane < 16) gh[nBase + nloc + lane] = acc[0] + bb;  // D row 0 = h . W^T

    grid_barrier(bar, GRU_BLOCKS);

    const float* gxt = gx + (size_t)t * H3;
    volatile const float* ghv = gh;
    for (int i = tid; i < HID; i += GRU_THREADS) {
      float r  = sigmoid_f(gxt[i]           + ghv[i]);
      float z  = sigmoid_f(gxt[i + HID]     + ghv[i + HID]);
      float n  = tanhf   (gxt[i + 2 * HID] + r * ghv[i + 2 * HID]);
      float hn = (1.0f - z) * n + z * sh[i];
      sh[i]  = hn;
      shb[i] = (__bf16)hn;
      if (blockIdx.x == 0) outs[(size_t)t * HID + i] = hn;
    }
    __syncthreads();
  }
}

// ---------------------------------------------------------------------
extern "C" void kernel_launch(void* const* d_in, const int* in_sizes, int n_in,
                              void* d_out, int out_size, void* d_ws, size_t ws_size,
                              hipStream_t stream) {
  const int*   loc        = (const int*)d_in[0];
  const int*   tim        = (const int*)d_in[1];
  const int*   clu        = (const int*)d_in[2];
  const int*   hloc       = (const int*)d_in[3];
  const int*   hclu       = (const int*)d_in[4];
  const int*   htim       = (const int*)d_in[5];
  const int*   hcnt       = (const int*)d_in[6];
  const int*   uid        = (const int*)d_in[7];
  /* d_in[8] = target_len (fixed T_LEN) */
  const float* emb_loc    = (const float*)d_in[9];
  const float* emb_tim    = (const float*)d_in[10];
  const float* emb_clu    = (const float*)d_in[11];
  const float* emb_uid    = (const float*)d_in[12];
  const float* fc_attn_w  = (const float*)d_in[13];
  const float* fc_attn_b  = (const float*)d_in[14];
  const float* w_ih       = (const float*)d_in[15];
  const float* w_hh       = (const float*)d_in[16];
  const float* b_ih       = (const float*)d_in[17];
  const float* b_hh       = (const float*)d_in[18];
  const float* fc_final_w = (const float*)d_in[19];
  const float* fc_final_b = (const float*)d_in[20];
  (void)in_sizes; (void)n_in; (void)out_size; (void)ws_size;

  unsigned char* ws = (unsigned char*)d_ws;
  size_t off = 0;
  auto take = [&](size_t bytes) -> unsigned char* {
    unsigned char* p = ws + off;
    off = (off + bytes + 255) & ~(size_t)255;
    return p;
  };
  int*   offs  = (int*)  take((size_t)G_CNT * 4);
  float* x     = (float*)take((size_t)S_LEN * INP * 4);
  float* hpre  = (float*)take((size_t)G_CNT * INP * 4);
  float* hist  = (float*)take((size_t)G_CNT * HID * 4);
  float* histT = (float*)take((size_t)HID * G_CNT * 4);
  float* gx    = (float*)take((size_t)S_LEN * H3 * 4);
  float* outs  = (float*)take((size_t)S_LEN * HID * 4);
  float* attn  = (float*)take((size_t)T_LEN * G_CNT * 4);
  float* ctx   = (float*)take((size_t)T_LEN * HID * 4);
  float* ocat  = (float*)take((size_t)T_LEN * CAT * 4);
  float* ghbuf = (float*)take((size_t)2 * H3 * 4);
  int*   bar   = (int*)  take(2 * 4);
  float* y     = (float*)d_out;                       // 512 x 50000 logits

  // 1) history offsets + feature gathering
  scan_offsets_kernel<<<1, G_CNT, 0, stream>>>(hcnt, offs);
  gather_x_kernel<<<S_LEN, 256, 0, stream>>>(loc, clu, tim, emb_loc, emb_clu, emb_tim, x);
  hist_feat_kernel<<<G_CNT, 256, 0, stream>>>(hloc, hclu, htim, hcnt, offs,
                                              emb_loc, emb_clu, emb_tim, hpre);
  // 2) history = tanh(hpre @ fc_attn_w^T + b)   (1024 x 512, K=608)
  gemm_nt_bf16<<<dim3(HID / 16, 2), 256, 0, stream>>>(hpre, fc_attn_w, fc_attn_b, hist,
                                                      G_CNT, HID, INP, 1);
  // 3) gx = x @ w_ih^T + b_ih                   (1024 x 1536, K=608)
  gemm_nt_bf16<<<dim3(H3 / 16, 2), 256, 0, stream>>>(x, w_ih, b_ih, gx,
                                                     S_LEN, H3, INP, 0);
  // 4) sequential GRU over 1024 steps, 16 WGPs cooperating
  ws_init_kernel<<<1, 32, 0, stream>>>(bar);
  gru_persistent<<<GRU_BLOCKS, GRU_THREADS, GRU_SMEM, stream>>>(gx, w_hh, b_hh,
                                                                outs, ghbuf, bar);
  // 5) attention over history
  const float* q = outs + (size_t)(S_LEN - T_LEN) * HID;
  gemm_nt_bf16<<<dim3(G_CNT / 16, 1), 256, 0, stream>>>(q, hist, nullptr, attn,
                                                        T_LEN, G_CNT, HID, 0);
  softmax_rows_kernel<<<T_LEN, 256, 0, stream>>>(attn, G_CNT);
  transpose_kernel<<<dim3(HID / 32, G_CNT / 32), 256, 0, stream>>>(hist, histT, G_CNT, HID);
  gemm_nt_bf16<<<dim3(HID / 16, 1), 256, 0, stream>>>(attn, histT, nullptr, ctx,
                                                      T_LEN, HID, G_CNT, 0);
  // 6) final projection: 55.7 GFLOP, 218MB weight streamed exactly once
  concat_out_kernel<<<T_LEN, 256, 0, stream>>>(q, ctx, emb_uid, uid, ocat);
  gemm_nt_bf16<<<dim3(LOC_N / 16, 1), 256, 0, stream>>>(ocat, fc_final_w, fc_final_b, y,
                                                        T_LEN, LOC_N, CAT, 0);
  log_softmax_rows_kernel<<<T_LEN, 256, 0, stream>>>(y, LOC_N);
}
